// BitConv2d_56075093017281
// MI455X (gfx1250) — 8-bit matrix operand order repaired and checked by reading, compiled, not run on hardware
//
#include <hip/hip_runtime.h>
#include <stdint.h>

typedef __attribute__((ext_vector_type(8))) int v8i_t;

#define N_IMG   16
#define C_IN    32
#define C_OUT   32
#define H_DIM   224
#define W_DIM   224
#define HP      226
#define WP      226
#define N_ELEM  (N_IMG * C_IN * H_DIM * W_DIM)
#define QX_BYTES ((size_t)N_IMG * HP * WP * C_IN)
#define W_TILES (W_DIM / 16)
#define M_TILES (N_IMG * H_DIM * W_TILES)
#define KTILES  5

__global__ __launch_bounds__(256) void absmax_kernel(const float4* __restrict__ x,
                                                     int n4, unsigned int* __restrict__ out) {
    float m = 0.0f;
    for (size_t i = (size_t)blockIdx.x * blockDim.x + threadIdx.x; i < (size_t)n4;
         i += (size_t)gridDim.x * blockDim.x) {
        float4 v = x[i];
        m = fmaxf(m, fmaxf(fmaxf(fabsf(v.x), fabsf(v.y)), fmaxf(fabsf(v.z), fabsf(v.w))));
    }
    __shared__ float red[256];
    red[threadIdx.x] = m;
    __syncthreads();
    for (int s = 128; s > 0; s >>= 1) {
        if ((int)threadIdx.x < s) red[threadIdx.x] = fmaxf(red[threadIdx.x], red[threadIdx.x + s]);
        __syncthreads();
    }
    if (threadIdx.x == 0) atomicMax(out, __float_as_uint(red[0]));
}

__global__ __launch_bounds__(256) void prep_kernel(const float* __restrict__ w,
                                                   const unsigned int* __restrict__ absmax_bits,
                                                   float* __restrict__ scale_out,
                                                   unsigned int* __restrict__ bfrag) {
    __shared__ float red[256];
    __shared__ float wscale_sh;
    float s = 0.0f;
    for (int i = threadIdx.x; i < C_OUT * C_IN * 9; i += 256) s += fabsf(w[i]);
    red[threadIdx.x] = s;
    __syncthreads();
    for (int st = 128; st > 0; st >>= 1) {
        if ((int)threadIdx.x < st) red[threadIdx.x] += red[threadIdx.x + st];
        __syncthreads();
    }
    if (threadIdx.x == 0) {
        float wsc = red[0] / (float)(C_OUT * C_IN * 9) + 1e-5f;
        wscale_sh = wsc;
        float xs = __uint_as_float(*absmax_bits) + 1e-5f;
        *scale_out = (xs / 127.0f) * wsc;
    }
    __syncthreads();
    float invw = 1.0f / wscale_sh;

    for (int idx = threadIdx.x; idx < KTILES * 2 * 32 * 8; idx += 256) {
        int v    = idx & 7;
        int lane = (idx >> 3) & 31;
        int nt   = (idx >> 8) & 1;
        int kt   = idx >> 9;
        int cout = nt * 16 + (lane & 15);
        int khalf = (lane < 16) ? 0 : 16;
        unsigned int dw = 0;
        for (int b = 0; b < 4; ++b) {
            int Klocal = (v >> 1) * 16 + (khalf >> 1) + (v & 1) * 4 + b;
            int K   = kt * 64 + Klocal;
            int tap = K >> 5;
            int cin = K & 31;
            int q = 0;
            if (tap < 9) {
                float r = rintf(w[(cout * C_IN + cin) * 9 + tap] * invw);
                q = (int)fminf(fmaxf(r, -1.0f), 1.0f);
            }
            dw |= ((unsigned int)(q & 0xFF)) << (8 * b);
        }
        bfrag[idx] = dw;
    }
}

__global__ __launch_bounds__(256) void quant_kernel(const float* __restrict__ x,
                                                    const unsigned int* __restrict__ absmax_bits,
                                                    uint8_t* __restrict__ qx) {
    __shared__ signed char sm[32][36];
    int b  = blockIdx.x;
    int wt = b % (W_DIM / 32);
    int r  = b / (W_DIM / 32);
    int h  = r % H_DIM;
    int n  = r / H_DIM;
    int w0 = wt * 32;

    float inv = 127.0f / (__uint_as_float(*absmax_bits) + 1e-5f);

    for (int i = 0; i < 4; ++i) {
        int t = threadIdx.x + i * 256;
        int c = t >> 5;
        int w = t & 31;
        float v = x[((((size_t)n * C_IN + c) * H_DIM + h) * W_DIM) + w0 + w];
        float q = rintf(fminf(fmaxf(v * inv, -128.0f), 127.0f));
        sm[w][c] = (signed char)(int)q;
    }
    __syncthreads();

    int w = threadIdx.x >> 3;
    int d = threadIdx.x & 7;
    unsigned int ow = ((unsigned int)(uint8_t)sm[w][d * 4 + 0]) |
                      ((unsigned int)(uint8_t)sm[w][d * 4 + 1] << 8) |
                      ((unsigned int)(uint8_t)sm[w][d * 4 + 2] << 16) |
                      ((unsigned int)(uint8_t)sm[w][d * 4 + 3] << 24);
    unsigned int* dst = (unsigned int*)(qx + (((size_t)n * HP + (h + 1)) * WP + (w0 + w + 1)) * C_IN);
    dst[d] = ow;
}

__global__ __launch_bounds__(256) void conv_kernel(const uint8_t* __restrict__ qx,
                                                   const unsigned int* __restrict__ bfrag,
                                                   const float* __restrict__ scale_p,
                                                   float* __restrict__ out) {
    __shared__ float sm[8][16][33];

    int wave = threadIdx.x >> 5;
    int lane = threadIdx.x & 31;
    int tile = blockIdx.x * 8 + wave;
    int n    = tile / (H_DIM * W_TILES);
    int rem  = tile % (H_DIM * W_TILES);
    int h    = rem / W_TILES;
    int w0   = (rem % W_TILES) * 16;

    int m    = lane & 15;
    int half = lane >> 4;

    v8i_t acc0 = {}, acc1 = {};

#pragma unroll
    for (int kt = 0; kt < KTILES; ++kt) {
        v8i_t a;
        {
            int t0 = 2 * kt;
            int dh = t0 / 3, dwp = t0 % 3;
            const uint8_t* p = qx + (((size_t)n * HP + (h + dh)) * WP + (w0 + m + dwp)) * C_IN
                                  + half * 8;
            uint2 lo = *(const uint2*)p;
            uint2 hi = *(const uint2*)(p + 16);
            a[0] = (int)lo.x; a[1] = (int)lo.y; a[2] = (int)hi.x; a[3] = (int)hi.y;
        }
        if (2 * kt + 1 < 9) {
            int t1 = 2 * kt + 1;
            int dh = t1 / 3, dwp = t1 % 3;
            const uint8_t* p = qx + (((size_t)n * HP + (h + dh)) * WP + (w0 + m + dwp)) * C_IN
                                  + half * 8;
            uint2 lo = *(const uint2*)p;
            uint2 hi = *(const uint2*)(p + 16);
            a[4] = (int)lo.x; a[5] = (int)lo.y; a[6] = (int)hi.x; a[7] = (int)hi.y;
        } else {
            a[4] = 0; a[5] = 0; a[6] = 0; a[7] = 0;
        }

        v8i_t b0 = *(const v8i_t*)(bfrag + (((kt * 2 + 0) * 32 + lane) << 3));
        v8i_t b1 = *(const v8i_t*)(bfrag + (((kt * 2 + 1) * 32 + lane) << 3));

        acc0 = __builtin_amdgcn_wmma_i32_16x16x64_iu8(true, a, true, b0, acc0, false, false);
        acc1 = __builtin_amdgcn_wmma_i32_16x16x64_iu8(true, a, true, b1, acc1, false, false);
    }

    float s = scale_p[0];
    int nidx = lane & 15;
#pragma unroll
    for (int r = 0; r < 8; ++r) {
        sm[wave][r + half * 8][nidx]      = (float)acc0[r] * s;
        sm[wave][r + half * 8][16 + nidx] = (float)acc1[r] * s;
    }
    __syncthreads();

#pragma unroll
    for (int p = 0; p < 16; ++p) {
        int cout = p * 2 + half;
        out[(((size_t)n * C_OUT + cout) * H_DIM + h) * W_DIM + w0 + m] = sm[wave][m][cout];
    }
}

extern "C" void kernel_launch(void* const* d_in, const int* in_sizes, int n_in,
                              void* d_out, int out_size, void* d_ws, size_t ws_size,
                              hipStream_t stream) {
    const float* x = (const float*)d_in[0];
    const float* w = (const float*)d_in[1];
    float* out = (float*)d_out;

    uint8_t* ws = (uint8_t*)d_ws;
    unsigned int* absmax_bits = (unsigned int*)ws;
    float*        scale       = (float*)(ws + 4);
    unsigned int* bfrag       = (unsigned int*)(ws + 256);
    uint8_t*      qx          = ws + 32768;

    hipMemsetAsync(absmax_bits, 0, sizeof(unsigned int), stream);
    hipMemsetAsync(qx, 0, QX_BYTES, stream);

    absmax_kernel<<<4096, 256, 0, stream>>>((const float4*)x, N_ELEM / 4, absmax_bits);
    prep_kernel<<<1, 256, 0, stream>>>(w, absmax_bits, scale, bfrag);
    quant_kernel<<<N_IMG * H_DIM * (W_DIM / 32), 256, 0, stream>>>(x, absmax_bits, qx);
    conv_kernel<<<M_TILES / 8, 256, 0, stream>>>(qx, bfrag, scale, out);
}
